// ParallelDroplessMLP_90125593739346
// MI455X (gfx1250) — compile-verified, matrix operand order checked by reading, not run on hardware
//
#include <hip/hip_runtime.h>
#include <hip/hip_bf16.h>

// ---- MoE dims (fixed by the reference) ----
#define T_TOK   4096
#define HID     1024
#define FFN_D   4096
#define NEXP    8
#define TOPK    2

// ---- fused-kernel tiling ----
#define M_BLK   32          // tokens per block
#define NWAVE   16          // waves per block (512 threads)
#define FC      256         // FFN chunk width  (= NWAVE * 16)
#define XS_STR  1032        // bf16 row stride for x tile   (2064B % 256 == 16)
#define HS_STR  264         // bf16 row stride for h chunk  (528B  % 256 == 16)

typedef __attribute__((ext_vector_type(16))) __bf16 v16bf;
typedef __attribute__((ext_vector_type(8)))  __bf16 v8bf;
typedef __attribute__((ext_vector_type(8)))  float  v8f;

union V16 { v16bf v; v8bf h[2]; };

__device__ __forceinline__ __bf16 f2bf(float f) {
    union { float f; unsigned u; } in; in.f = f;
    unsigned r = (in.u + 0x7FFFu + ((in.u >> 16) & 1u)) >> 16;  // RNE
    union { unsigned short s; __bf16 b; } out; out.s = (unsigned short)r;
    return out.b;
}

// ---------------------------------------------------------------------------
// Pack pass 1: batched transpose + f32->bf16.  in: [B][R][C] f32 row-major,
// out: [B][C][R] bf16 (K contiguous per output column for WMMA B-fragments).
// ---------------------------------------------------------------------------
__global__ void transpose_pack_kernel(const float* __restrict__ in,
                                      __bf16* __restrict__ out,
                                      int R, int C) {
    __shared__ float tile[32][33];
    const int b  = blockIdx.z;
    const float* inb  = in  + (size_t)b * R * C;
    __bf16*      outb = out + (size_t)b * R * C;

    const int cx = blockIdx.x * 32 + threadIdx.x;
    const int r0 = blockIdx.y * 32;
#pragma unroll
    for (int jj = 0; jj < 4; ++jj) {
        int r = r0 + threadIdx.y + jj * 8;
        tile[threadIdx.y + jj * 8][threadIdx.x] = inb[(size_t)r * C + cx];
    }
    __syncthreads();
    const int co = blockIdx.x * 32;
    const int ro = r0 + threadIdx.x;
#pragma unroll
    for (int jj = 0; jj < 4; ++jj) {
        outb[(size_t)(co + threadIdx.y + jj * 8) * R + ro] =
            f2bf(tile[threadIdx.x][threadIdx.y + jj * 8]);
    }
}

// ---------------------------------------------------------------------------
// Pack pass 2: dense gate table  gate[t][e] = sum_k (idx[t][k]==e) * w[t][k]
// ---------------------------------------------------------------------------
__global__ void gates_kernel(const int* __restrict__ idx,
                             const float* __restrict__ w,
                             float* __restrict__ gates) {
    int t = blockIdx.x * blockDim.x + threadIdx.x;
    if (t >= T_TOK) return;
    float g[NEXP];
#pragma unroll
    for (int i = 0; i < NEXP; ++i) g[i] = 0.0f;
#pragma unroll
    for (int k = 0; k < TOPK; ++k) {
        int e = idx[t * TOPK + k] & (NEXP - 1);
        g[e] += w[t * TOPK + k];
    }
#pragma unroll
    for (int i = 0; i < NEXP; ++i) gates[(size_t)t * NEXP + i] = g[i];
}

// ---------------------------------------------------------------------------
// Fused MoE: block = 32 tokens, 16 waves.  Wave w owns output cols
// [w*64, w*64+64).  Each B fragment feeds TWO row-tiles (32 FLOP/B from L2).
// For each expert: FFN streamed in 256-wide chunks: GEMM1 (K=1024) ->
// gelu, routing gate folded in -> bf16 LDS -> GEMM2 (K=256) accumulating
// gate*y into persistent f32 WMMA accumulators across all experts.
// ---------------------------------------------------------------------------
__global__ __launch_bounds__(NWAVE * 32)
void moe_fused_kernel(const float* __restrict__ x,
                      const float* __restrict__ gates,
                      const __bf16* __restrict__ w1t,   // [E][FFN][HID]
                      const __bf16* __restrict__ w2t,   // [E][HID][FFN]
                      float* __restrict__ out) {
    __shared__ __bf16 xs[M_BLK * XS_STR];   // 66 KB
    __shared__ __bf16 hs[M_BLK * HS_STR];   // 16.5 KB

    const int tid  = threadIdx.x;
    const int wave = tid >> 5;
    const int lane = tid & 31;
    const int half = lane >> 4;        // 0: lanes 0-15, 1: lanes 16-31
    const int l16  = lane & 15;
    const int t0   = blockIdx.x * M_BLK;

    // ---- load x tile [32 x 1024] f32 -> bf16 LDS (coalesced float4) ----
#pragma unroll
    for (int j = 0; j < (M_BLK * HID / 4) / (NWAVE * 32); ++j) {
        int elem = (tid + j * NWAVE * 32) * 4;
        int row = elem >> 10, col = elem & 1023;
        const float4 v = *(const float4*)(x + (size_t)(t0 + row) * HID + col);
        __bf16* p = &xs[row * XS_STR + col];
        p[0] = f2bf(v.x); p[1] = f2bf(v.y); p[2] = f2bf(v.z); p[3] = f2bf(v.w);
    }
    __syncthreads();

    // persistent output accumulators: 2 row-tiles x 4 col-tiles of 16x16
    v8f oacc0[4], oacc1[4];
#pragma unroll
    for (int j = 0; j < 4; ++j)
#pragma unroll
        for (int i = 0; i < 8; ++i) { oacc0[j][i] = 0.0f; oacc1[j][i] = 0.0f; }

    for (int e = 0; e < NEXP; ++e) {
        float gA[8], gB[8];
#pragma unroll
        for (int i = 0; i < 8; ++i) {
            gA[i] = gates[(size_t)(t0 + i + half * 8) * NEXP + e];
            gB[i] = gates[(size_t)(t0 + 16 + i + half * 8) * NEXP + e];
        }

        for (int f0 = 0; f0 < FFN_D; f0 += FC) {
            // ---------- GEMM1: h[32 x 16] for cols f0 + wave*16 ----------
            v8f hacc0, hacc1;
#pragma unroll
            for (int i = 0; i < 8; ++i) { hacc0[i] = 0.0f; hacc1[i] = 0.0f; }

            const int nb = f0 + wave * 16 + l16;
            const __bf16* bptr = w1t + ((size_t)e * FFN_D + nb) * HID + half * 8;
            const __bf16* aptr = xs + l16 * XS_STR + half * 8;
#pragma unroll 4
            for (int k0 = 0; k0 < HID; k0 += 32) {
                V16 A0, A1, B;
                B.h[0]  = *(const v8bf*)(bptr + k0);
                B.h[1]  = *(const v8bf*)(bptr + k0 + 16);
                A0.h[0] = *(const v8bf*)(aptr + k0);
                A0.h[1] = *(const v8bf*)(aptr + k0 + 16);
                A1.h[0] = *(const v8bf*)(aptr + 16 * XS_STR + k0);
                A1.h[1] = *(const v8bf*)(aptr + 16 * XS_STR + k0 + 16);
                __builtin_prefetch(bptr + k0 + 128, 0, 0);
                hacc0 = __builtin_amdgcn_wmma_f32_16x16x32_bf16(
                    false, A0.v, false, B.v, (short)0, hacc0, false, false);
                hacc1 = __builtin_amdgcn_wmma_f32_16x16x32_bf16(
                    false, A1.v, false, B.v, (short)0, hacc1, false, false);
            }

            __syncthreads();   // prior chunk's GEMM2 reads of hs are done
            // ---- exact gelu, fold routing gate, write bf16 h chunk ----
            {
                const int col = wave * 16 + l16;
#pragma unroll
                for (int i = 0; i < 8; ++i) {
                    float v0 = hacc0[i], v1 = hacc1[i];
                    float g0 = 0.5f * v0 * (1.0f + erff(v0 * 0.70710678118f));
                    float g1 = 0.5f * v1 * (1.0f + erff(v1 * 0.70710678118f));
                    hs[(i + half * 8) * HS_STR + col]        = f2bf(g0 * gA[i]);
                    hs[(16 + i + half * 8) * HS_STR + col]   = f2bf(g1 * gB[i]);
                }
            }
            __syncthreads();

            // ---------- GEMM2: oacc += h_chunk @ W2[f0:f0+FC, :] ----------
            const __bf16* a2 = hs + l16 * HS_STR + half * 8;
#pragma unroll
            for (int j = 0; j < 4; ++j) {
                const int n2 = wave * 64 + j * 16 + l16;
                const __bf16* b2 =
                    w2t + ((size_t)e * HID + n2) * FFN_D + f0 + half * 8;
                v8f acc0 = oacc0[j], acc1 = oacc1[j];
#pragma unroll 4
                for (int k0 = 0; k0 < FC; k0 += 32) {
                    V16 A0, A1, B;
                    B.h[0]  = *(const v8bf*)(b2 + k0);
                    B.h[1]  = *(const v8bf*)(b2 + k0 + 16);
                    A0.h[0] = *(const v8bf*)(a2 + k0);
                    A0.h[1] = *(const v8bf*)(a2 + k0 + 16);
                    A1.h[0] = *(const v8bf*)(a2 + 16 * HS_STR + k0);
                    A1.h[1] = *(const v8bf*)(a2 + 16 * HS_STR + k0 + 16);
                    acc0 = __builtin_amdgcn_wmma_f32_16x16x32_bf16(
                        false, A0.v, false, B.v, (short)0, acc0, false, false);
                    acc1 = __builtin_amdgcn_wmma_f32_16x16x32_bf16(
                        false, A1.v, false, B.v, (short)0, acc1, false, false);
                }
                oacc0[j] = acc0; oacc1[j] = acc1;
            }
        }
    }

    // ---- store f32 output (every (t,h) covered exactly once) ----
#pragma unroll
    for (int j = 0; j < 4; ++j) {
        const int col = wave * 64 + j * 16 + l16;
#pragma unroll
        for (int i = 0; i < 8; ++i) {
            out[(size_t)(t0 + i + half * 8) * HID + col]      = oacc0[j][i];
            out[(size_t)(t0 + 16 + i + half * 8) * HID + col] = oacc1[j][i];
        }
    }
}

// ---------------------------------------------------------------------------
extern "C" void kernel_launch(void* const* d_in, const int* in_sizes, int n_in,
                              void* d_out, int out_size, void* d_ws, size_t ws_size,
                              hipStream_t stream) {
    const float* x  = (const float*)d_in[0];
    // d_in[1] = scores: unused by the reference
    const float* ew = (const float*)d_in[2];
    const int*   ei = (const int*)d_in[3];
    const float* w1 = (const float*)d_in[4];
    const float* w2 = (const float*)d_in[5];
    float* out = (float*)d_out;

    const size_t wbytes = (size_t)NEXP * HID * FFN_D * sizeof(__bf16); // 64 MiB
    __bf16* w1t   = (__bf16*)d_ws;                                     // [E][F][H]
    __bf16* w2t   = (__bf16*)((char*)d_ws + wbytes);                   // [E][H][F]
    float*  gates = (float*)((char*)d_ws + 2 * wbytes);                // [T][E]

    transpose_pack_kernel<<<dim3(FFN_D / 32, HID / 32, NEXP), dim3(32, 8), 0, stream>>>(
        w1, w1t, HID, FFN_D);
    transpose_pack_kernel<<<dim3(HID / 32, FFN_D / 32, NEXP), dim3(32, 8), 0, stream>>>(
        w2, w2t, FFN_D, HID);
    gates_kernel<<<(T_TOK + 255) / 256, 256, 0, stream>>>(ei, ew, gates);

    moe_fused_kernel<<<T_TOK / M_BLK, NWAVE * 32, 0, stream>>>(x, gates, w1t, w2t, out);
}
